// WindowAttention3D_44710609551872
// MI455X (gfx1250) — matrix operand order repaired and checked by reading, not compiled, not run on hardware
//
#include <hip/hip_runtime.h>
#include <hip/hip_bf16.h>
#include <math.h>

typedef __attribute__((ext_vector_type(16))) _Float16 v16h;
typedef __attribute__((ext_vector_type(8)))  _Float16 v8h;
typedef __attribute__((ext_vector_type(8)))  float    v8f;
typedef __attribute__((ext_vector_type(4)))  int      v4i;

#define TT    6
#define HWS   64
#define NN    384        // TT*HWS
#define DIMC  192
#define NHC   6
#define HDC   32
#define BBATCH 128
#define SCALE_Q 0.17677669529663689f  // 1/sqrt(32)

#define GLOBAL_AS __attribute__((address_space(1)))
#define LDS_AS    __attribute__((address_space(3)))

#if __has_builtin(__builtin_amdgcn_global_load_async_to_lds_b128) && \
    __has_builtin(__builtin_amdgcn_s_wait_asynccnt)
#define HAVE_ASYNC_LDS 1
#endif

// 16-byte copy global -> LDS; async (ASYNCcnt) when the gfx1250 builtins exist.
static __device__ __forceinline__ void copy_b128_to_lds(void* lds_dst, const void* gsrc) {
#ifdef HAVE_ASYNC_LDS
  __builtin_amdgcn_global_load_async_to_lds_b128(
      (GLOBAL_AS v4i*)gsrc, (LDS_AS v4i*)lds_dst, 0, 0);
#else
  *(v8h*)lds_dst = *(const v8h*)gsrc;
#endif
}

static __device__ __forceinline__ void wait_lds_copies() {
#ifdef HAVE_ASYNC_LDS
  __builtin_amdgcn_s_wait_asynccnt(0);
#endif
}

// ---------------- WMMA fragment helpers (CDNA5 16x16x32 f16 layouts) ----------------

static __device__ __forceinline__ v8f vzero() {
  v8f z;
#pragma unroll
  for (int e = 0; e < 8; ++e) z[e] = 0.0f;
  return z;
}

// A-matrix 16x32 f16: lane L -> row m = L&15; lanes 0-15 hold K {0..7,16..23},
// lanes 16-31 hold K {8..15,24..31}. Two contiguous 8-half (16B) chunks per lane.
static __device__ __forceinline__ v16h make_afrag(const _Float16* base, int ld,
                                                  int m0, int k0, int lane) {
  int m  = m0 + (lane & 15);
  int hh = lane >> 4;
  const _Float16* p = base + (size_t)m * ld + k0 + 8 * hh;
  v8h lo = *(const v8h*)p;
  v8h hi = *(const v8h*)(p + 16);
  v16h r;
#pragma unroll
  for (int e = 0; e < 8; ++e) { r[e] = lo[e]; r[8 + e] = hi[e]; }
  return r;
}

// B-matrix 32x16 f16 stored column-major-in-K (src[n*ld + k]): lane n<16 holds
// K {0..7,16..23} of column n; lane n+16 holds K {8..15,24..31}. Two 8-half chunks.
static __device__ __forceinline__ v16h make_bfrag(const _Float16* base, int ld,
                                                  int n0, int k0, int lane) {
  int n  = n0 + (lane & 15);
  int hh = lane >> 4;
  const _Float16* p = base + (size_t)n * ld + k0 + 8 * hh;
  v8h lo = *(const v8h*)p;
  v8h hi = *(const v8h*)(p + 16);
  v16h r;
#pragma unroll
  for (int e = 0; e < 8; ++e) { r[e] = lo[e]; r[8 + e] = hi[e]; }
  return r;
}

static __device__ __forceinline__ v8f wmma_f16(v16h a, v16h b, v8f c) {
  return __builtin_amdgcn_wmma_f32_16x16x32_f16(false, a, false, b,
                                                (short)0, c, false, false);
}

// ---------------- Kernel 1: positional encoding + weight f16 conversion ----------------

__global__ void prep_kernel(const float* __restrict__ Wqkv, const float* __restrict__ Wproj,
                            const float* __restrict__ W1,   const float* __restrict__ W2,
                            float* __restrict__ posf, _Float16* __restrict__ wq16,
                            _Float16* __restrict__ wp16, _Float16* __restrict__ w1h,
                            _Float16* __restrict__ w2h) {
  int idx = blockIdx.x * blockDim.x + threadIdx.x;
  const int S0 = HWS * DIMC;              // 12288
  const int S1 = S0 + 3 * DIMC * DIMC;    // +110592
  const int S2 = S1 + DIMC * 2 * DIMC;    // +73728
  const int S3 = S2 + DIMC * DIMC;
  const int S4 = S3 + DIMC * DIMC;
  if (idx < S0) {
    int p = idx / DIMC, c = idx % DIMC;
    int py = p >> 3, px = p & 7;
    const float twopi = 6.28318530717958647692f;
    float embed = (c < 96) ? ((float)(py + 1) / (8.0f + 1e-6f) * twopi)
                           : ((float)(px + 1) / (8.0f + 1e-6f) * twopi);
    int cc = (c < 96) ? c : (c - 96);
    float dt = powf(10000.0f, (float)(cc & ~1) / 96.0f);
    float v = embed / dt;
    posf[idx] = (cc & 1) ? cosf(v) : sinf(v);
  } else if (idx < S1) { int i = idx - S0; wq16[i] = (_Float16)Wqkv[i]; }
  else if (idx < S2)   { int i = idx - S1; wp16[i] = (_Float16)Wproj[i]; }
  else if (idx < S3)   { int i = idx - S2; w1h[i]  = (_Float16)W1[i]; }
  else if (idx < S4)   { int i = idx - S3; w2h[i]  = (_Float16)W2[i]; }
}

// ---------------- Kernel 2: QKV GEMM (WMMA), scatter into head-major f16 ----------------

__global__ __launch_bounds__(256) void qkv_kernel(const float* __restrict__ x,
                                                  const float* __restrict__ posf,
                                                  const _Float16* __restrict__ wq16,
                                                  _Float16* __restrict__ qws,
                                                  _Float16* __restrict__ kws,
                                                  _Float16* __restrict__ vws) {
  __shared__ _Float16 xa[64 * DIMC];
  const int tid  = threadIdx.x;
  const int lane = tid & 31;
  const int wave = tid >> 5;
  const int wg = blockIdx.x;               // 768 blocks, 64 rows each
  const int bb = (wg * 64) / NN;
  const int nbase = (wg * 64) % NN;        // 64-aligned -> one frame per WG
  const int tf = nbase >> 6;               // frame index, uniform per WG

  for (int idx = tid; idx < 64 * DIMC; idx += 256) {
    int m = idx / DIMC, c = idx % DIMC;
    float v = x[((size_t)bb * NN + nbase + m) * DIMC + c] + posf[m * DIMC + c];
    xa[m * DIMC + c] = (_Float16)v;
  }
  __syncthreads();

  // 4 m-blocks x 36 n-blocks = 144 tiles, 18 per wave; K = 192 = 6 k-steps
  for (int tile = wave; tile < 144; tile += 8) {
    int mb = tile & 3, nb = tile >> 2;
    v8f acc = vzero();
#pragma unroll
    for (int kk = 0; kk < 6; ++kk) {
      v16h a = make_afrag(xa, DIMC, mb * 16, kk * 32, lane);
      v16h b = make_bfrag(wq16, DIMC, nb * 16, kk * 32, lane);
      acc = wmma_f16(a, b, acc);
    }
    // tile-uniform decomposition: q/k/v segment, head, d-half
    int s  = nb / 12;                       // 0:q 1:k 2:v (wave-uniform)
    int nr = nb - s * 12;
    int hd = nr >> 1;                       // head (uniform)
    int d  = (nr & 1) * 16 + (lane & 15);   // channel within head
    int r0 = mb * 16 + 8 * (lane >> 4);     // key/query position base
    size_t base = (((size_t)bb * NHC + hd) * TT + tf);
    if (s == 0) {
      _Float16* qp = qws + (base * HWS + r0) * HDC + d;
#pragma unroll
      for (int g = 0; g < 8; ++g) qp[g * HDC] = (_Float16)(acc[g] * SCALE_Q);
    } else if (s == 1) {
      _Float16* kp = kws + (base * HWS + r0) * HDC + d;
#pragma unroll
      for (int g = 0; g < 8; ++g) kp[g * HDC] = (_Float16)acc[g];
    } else {
      // transposed [d][p]: the 8 elements are consecutive p -> one 16B store
      v8h pk;
#pragma unroll
      for (int g = 0; g < 8; ++g) pk[g] = (_Float16)acc[g];
      *(v8h*)(vws + (base * HDC + d) * HWS + r0) = pk;
    }
  }
}

// ---------------- Kernel 3: fused attention + proj + LN + MLP, one WG per (i, b) ----------------

__global__ __launch_bounds__(256) void fused_kernel(const float* __restrict__ x,
                                                    const _Float16* __restrict__ qws,
                                                    const _Float16* __restrict__ kws,
                                                    const _Float16* __restrict__ vws,
                                                    const _Float16* __restrict__ wp16,
                                                    const float* __restrict__ bproj,
                                                    const float* __restrict__ gamma,
                                                    const float* __restrict__ beta,
                                                    const _Float16* __restrict__ w1h,
                                                    const float* __restrict__ b1,
                                                    const _Float16* __restrict__ w2h,
                                                    const float* __restrict__ b2,
                                                    float* __restrict__ out) {
  extern __shared__ char smem[];
  _Float16* qs  = (_Float16*)smem;            // 6*64*32 = 12288 halves
  _Float16* ks  = qs + 12288;                 // 12288 halves
  _Float16* vs  = ks + 12288;                 // 12288 halves ([head][d][key])
  _Float16* xa  = vs + 12288;                 // 64*384 halves: [o | x_i]
  float*    ybuf = (float*)(xa + 64 * 2 * DIMC);  // 64*192 f32
  float*    sc  = ybuf;                       // overlay: 64*64 scores
  _Float16* at  = (_Float16*)(sc + HWS * HWS);// overlay: 64*64 attn f16
  _Float16* ya  = vs;                         // overlay post-attention
  _Float16* hb  = ks;                         // overlay post-attention

  const int tid  = threadIdx.x;
  const int lane = tid & 31;
  const int wave = tid >> 5;
  const int i  = blockIdx.x;                  // query frame
  const int bb = blockIdx.y;                  // batch

  // Warm L2 for the streamed f16 weights.
  {
    const char* w = (const char*)wp16;
    for (int t3 = tid; t3 < (2 * DIMC * DIMC * 2) / 128; t3 += 256)
      __builtin_prefetch(w + t3 * 128, 0, 1);
    const char* w1p = (const char*)w1h;
    const char* w2p = (const char*)w2h;
    for (int t3 = tid; t3 < (DIMC * DIMC * 2) / 128; t3 += 256) {
      __builtin_prefetch(w1p + t3 * 128, 0, 1);
      __builtin_prefetch(w2p + t3 * 128, 0, 1);
    }
  }

  // Load q_i (all heads, async-to-LDS) and x_i once.
#pragma unroll
  for (int h = 0; h < NHC; ++h) {
    const _Float16* srcq = qws + (((size_t)bb * NHC + h) * TT + i) * 2048;
    copy_b128_to_lds(qs + h * 2048 + tid * 8, srcq + tid * 8);  // 2048 halves = 256 chunks
  }
  const float* xrow = x + (((size_t)i * BBATCH + bb) * HWS) * DIMC;  // x_tmp[i, bb]
  for (int idx = tid; idx < HWS * DIMC; idx += 256) {
    int m = idx / DIMC, c = idx % DIMC;
    xa[m * 2 * DIMC + DIMC + c] = (_Float16)xrow[idx];
  }

  v8f acc2[6];
#pragma unroll
  for (int s = 0; s < 6; ++s) acc2[s] = vzero();

  for (int j = 0; j < TT; ++j) {
    __syncthreads();   // previous iteration done reading ks/vs overlays
#pragma unroll
    for (int h = 0; h < NHC; ++h) {
      const _Float16* srck = kws + (((size_t)bb * NHC + h) * TT + j) * 2048;
      const _Float16* srcv = vws + (((size_t)bb * NHC + h) * TT + j) * 2048;
      copy_b128_to_lds(ks + h * 2048 + tid * 8, srck + tid * 8);
      copy_b128_to_lds(vs + h * 2048 + tid * 8, srcv + tid * 8);
    }
    wait_lds_copies();       // s_wait_asynccnt 0 (covers q on first iteration)
    __syncthreads();

    // ---- attention, head by head ----
    for (int h = 0; h < NHC; ++h) {
      const _Float16* qh = qs + h * 2048;
      const _Float16* kh = ks + h * 2048;
      const _Float16* vh = vs + h * 2048;
      // scores = q @ k^T : 16 tiles, 2 per wave, K=32 (one WMMA each)
#pragma unroll
      for (int t2 = 0; t2 < 2; ++t2) {
        int tile = wave * 2 + t2, mb = tile >> 2, nb = tile & 3;
        v16h a = make_afrag(qh, HDC, mb * 16, 0, lane);
        v16h b = make_bfrag(kh, HDC, nb * 16, 0, lane);
        v8f c = vzero();
        c = wmma_f16(a, b, c);
        int c0 = nb * 16 + (lane & 15);
        int r0 = mb * 16 + 8 * (lane >> 4);
#pragma unroll
        for (int g = 0; g < 8; ++g) sc[(r0 + g) * HWS + c0] = c[g];
      }
      __syncthreads();
      // softmax over keys (64), one row per lane
      if (tid < HWS) {
        const float* row = sc + tid * HWS;
        float mx = row[0];
        for (int c2 = 1; c2 < HWS; ++c2) mx = fmaxf(mx, row[c2]);
        float sum = 0.0f;
        for (int c2 = 0; c2 < HWS; ++c2) sum += __expf(row[c2] - mx);
        float inv = 1.0f / sum;
        _Float16* arow = at + tid * HWS;
        for (int c2 = 0; c2 < HWS; ++c2)
          arow[c2] = (_Float16)(__expf(row[c2] - mx) * inv);
      }
      __syncthreads();
      // o = attn @ v : 8 tiles, 1 per wave, K=64 (2 WMMA)
      {
        int mb = wave >> 1, nb = wave & 1;
        v8f c = vzero();
#pragma unroll
        for (int kk = 0; kk < 2; ++kk) {
          v16h a = make_afrag(at, HWS, mb * 16, kk * 32, lane);
          v16h b = make_bfrag(vh, HWS, nb * 16, kk * 32, lane);
          c = wmma_f16(a, b, c);
        }
        int c0 = h * HDC + nb * 16 + (lane & 15);
        int r0 = mb * 16 + 8 * (lane >> 4);
#pragma unroll
        for (int g = 0; g < 8; ++g)
          xa[(r0 + g) * 2 * DIMC + c0] = (_Float16)c[g];
      }
      __syncthreads();
    }

    // ---- y = [o | x_i] @ Wproj^T + bproj : 48 tiles, 6/wave, K=384 ----
    for (int s = 0; s < 6; ++s) {
      int tile = wave + 8 * s, mb = tile & 3, nb = tile >> 2;
      v8f c = vzero();
#pragma unroll
      for (int kk = 0; kk < 12; ++kk) {
        v16h a = make_afrag(xa, 2 * DIMC, mb * 16, kk * 32, lane);
        v16h b = make_bfrag(wp16, 2 * DIMC, nb * 16, kk * 32, lane);
        c = wmma_f16(a, b, c);
      }
      int c0 = nb * 16 + (lane & 15);
      int r0 = mb * 16 + 8 * (lane >> 4);
      float bias = bproj[c0];
#pragma unroll
      for (int g = 0; g < 8; ++g) ybuf[(r0 + g) * DIMC + c0] = c[g] + bias;
    }
    __syncthreads();

    // ---- LayerNorm, one row per lane ----
    if (tid < HWS) {
      const float* row = ybuf + tid * DIMC;
      float mu = 0.0f;
      for (int c2 = 0; c2 < DIMC; ++c2) mu += row[c2];
      mu *= (1.0f / DIMC);
      float var = 0.0f;
      for (int c2 = 0; c2 < DIMC; ++c2) { float d0 = row[c2] - mu; var += d0 * d0; }
      var *= (1.0f / DIMC);
      float inv = rsqrtf(var + 1e-5f);
      _Float16* yr = ya + tid * DIMC;
      for (int c2 = 0; c2 < DIMC; ++c2)
        yr[c2] = (_Float16)((row[c2] - mu) * inv * gamma[c2] + beta[c2]);
    }
    __syncthreads();

    // ---- h = gelu(y @ W1^T + b1) : 48 tiles, K=192 ----
    for (int s = 0; s < 6; ++s) {
      int tile = wave + 8 * s, mb = tile & 3, nb = tile >> 2;
      v8f c = vzero();
#pragma unroll
      for (int kk = 0; kk < 6; ++kk) {
        v16h a = make_afrag(ya, DIMC, mb * 16, kk * 32, lane);
        v16h b = make_bfrag(w1h, DIMC, nb * 16, kk * 32, lane);
        c = wmma_f16(a, b, c);
      }
      int c0 = nb * 16 + (lane & 15);
      int r0 = mb * 16 + 8 * (lane >> 4);
      float bias = b1[c0];
#pragma unroll
      for (int g = 0; g < 8; ++g) {
        float v0 = c[g] + bias;
        v0 = 0.5f * v0 * (1.0f + erff(v0 * 0.70710678118654752f));  // exact GELU
        hb[(r0 + g) * DIMC + c0] = (_Float16)v0;
      }
    }
    __syncthreads();

    // ---- y2 = h @ W2^T, accumulated over j in registers ----
    for (int s = 0; s < 6; ++s) {
      int tile = wave + 8 * s, mb = tile & 3, nb = tile >> 2;
      v8f c = acc2[s];
#pragma unroll
      for (int kk = 0; kk < 6; ++kk) {
        v16h a = make_afrag(hb, DIMC, mb * 16, kk * 32, lane);
        v16h b = make_bfrag(w2h, DIMC, nb * 16, kk * 32, lane);
        c = wmma_f16(a, b, c);
      }
      acc2[s] = c;
    }
  }

  // Epilogue: x_out = x + (sum_j gemm)/T + b2 -> out[..., DIM:2*DIM]
  const float* xsrc = x + ((size_t)bb * NN + (size_t)i * HWS) * DIMC;
  float* outp = out + ((size_t)bb * NN + (size_t)i * HWS) * (2 * DIMC);
  for (int s = 0; s < 6; ++s) {
    int tile = wave + 8 * s, mb = tile & 3, nb = tile >> 2;
    int c0 = nb * 16 + (lane & 15);
    int r0 = mb * 16 + 8 * (lane >> 4);
    float bias = b2[c0];
#pragma unroll
    for (int g = 0; g < 8; ++g) {
      int r = r0 + g;
      float xv = xsrc[(size_t)r * DIMC + c0];
      outp[(size_t)r * 2 * DIMC + DIMC + c0] = xv + acc2[s][g] * (1.0f / 6.0f) + bias;
    }
  }
}

// ---------------- Kernel 4: copy x into out[..., 0:DIM] ----------------

__global__ void copyx_kernel(const float* __restrict__ x, float* __restrict__ out) {
  size_t idx = (size_t)blockIdx.x * blockDim.x + threadIdx.x;  // over B*N*DIM
  size_t row = idx / DIMC;
  int c = (int)(idx % DIMC);
  out[row * 2 * DIMC + c] = x[idx];
}

// ---------------- Launch ----------------

extern "C" void kernel_launch(void* const* d_in, const int* in_sizes, int n_in,
                              void* d_out, int out_size, void* d_ws, size_t ws_size,
                              hipStream_t stream) {
  const float* x     = (const float*)d_in[0];
  const float* Wqkv  = (const float*)d_in[1];
  const float* Wproj = (const float*)d_in[2];
  const float* bproj = (const float*)d_in[3];
  const float* gamma = (const float*)d_in[4];
  const float* beta  = (const float*)d_in[5];
  const float* W1    = (const float*)d_in[6];
  const float* b1    = (const float*)d_in[7];
  const float* W2    = (const float*)d_in[8];
  const float* b2    = (const float*)d_in[9];
  float* out = (float*)d_out;

  char* ws = (char*)d_ws;
  size_t off = 0;
  auto carve = [&](size_t bytes) -> char* {
    char* p = ws + off;
    off += (bytes + 255) & ~(size_t)255;
    return p;
  };
  float*    posf = (float*)carve((size_t)HWS * DIMC * sizeof(float));
  _Float16* wq16 = (_Float16*)carve((size_t)3 * DIMC * DIMC * 2);
  _Float16* wp16 = (_Float16*)carve((size_t)DIMC * 2 * DIMC * 2);
  _Float16* w1h  = (_Float16*)carve((size_t)DIMC * DIMC * 2);
  _Float16* w2h  = (_Float16*)carve((size_t)DIMC * DIMC * 2);
  const size_t qkv_elems = (size_t)BBATCH * NHC * TT * HWS * HDC;   // 9,437,184
  _Float16* qws = (_Float16*)carve(qkv_elems * 2);
  _Float16* kws = (_Float16*)carve(qkv_elems * 2);
  _Float16* vws = (_Float16*)carve(qkv_elems * 2);

  const int prep_total = HWS * DIMC + 3 * DIMC * DIMC + DIMC * 2 * DIMC + 2 * DIMC * DIMC;
  prep_kernel<<<(prep_total + 255) / 256, 256, 0, stream>>>(
      Wqkv, Wproj, W1, W2, posf, wq16, wp16, w1h, w2h);

  qkv_kernel<<<(BBATCH * NN) / 64, 256, 0, stream>>>(x, posf, wq16, qws, kws, vws);

  copyx_kernel<<<(int)(((size_t)BBATCH * NN * DIMC) / 256), 256, 0, stream>>>(x, out);

  const size_t smem = (size_t)(12288 * 3 + 64 * 2 * DIMC) * 2 + (size_t)64 * DIMC * 4;  // 172032 B
  (void)hipFuncSetAttribute((const void*)fused_kernel,
                            hipFuncAttributeMaxDynamicSharedMemorySize, (int)smem);
  fused_kernel<<<dim3(TT, BBATCH), 256, smem, stream>>>(
      x, qws, kws, vws, wp16, bproj, gamma, beta, w1h, b1, w2h, b2, out);
}
